// Transformer_24421184045444
// MI455X (gfx1250) — matrix operand order repaired and checked by reading, not compiled, not run on hardware
//
#include <hip/hip_runtime.h>
#include <hip/hip_bf16.h>

// ---------------------------------------------------------------------------
// Transformer forward (B=4,S=1024,D=1024,H=8,E=1024,L=2,V=32000) for gfx1250.
// All GEMMs via v_wmma_f32_16x16x32_bf16 (f32 accumulate). B-operand tiles are
// staged global->LDS with the CDNA5 async-DMA path (ASYNCcnt), double-buffered
// so the next tile's copy overlaps the current tile's WMMAs; A streams through
// VGPRs with global_prefetch ahead. Weights converted+transposed to bf16 once
// per launch. Workspace need: ~640 MB.
// ---------------------------------------------------------------------------

typedef __bf16 bf16;
typedef __attribute__((ext_vector_type(16))) __bf16 v16bf;
typedef __attribute__((ext_vector_type(8)))  __bf16 v8bf;
typedef __attribute__((ext_vector_type(8)))  float  v8f;
typedef __attribute__((ext_vector_type(4)))  int    v4i;

typedef __attribute__((address_space(1))) v4i gv4i;   // global 16B chunk
typedef __attribute__((address_space(3))) v4i sv4i;   // LDS 16B chunk

static constexpr int Bq = 4;        // batch
static constexpr int Sq = 1024;     // seq len
static constexpr int Dq = 1024;     // d_model
static constexpr int Hq = 8;        // heads
static constexpr int Eq = 1024;     // head dim (QS == d_model)
static constexpr int Lq = 2;        // layers
static constexpr int Vq = 32000;    // vocab
static constexpr int Fq = 4096;     // FFN hidden

enum { FLAG_RELU = 1, FLAG_BF16 = 2, FLAG_TRANS = 4 };

// ---- CDNA5 async global->LDS copy (16B per lane), ASYNCcnt tracked ---------
__device__ inline void async_copy_b128(const bf16* g, bf16* l) {
#if __has_builtin(__builtin_amdgcn_global_load_async_to_lds_b128)
  __builtin_amdgcn_global_load_async_to_lds_b128(
      (gv4i*)(void*)g, (sv4i*)(void*)l, 0, 0);
#else
  unsigned lds_off = (unsigned)(unsigned long long)(sv4i*)(void*)l;
  asm volatile("global_load_async_to_lds_b128 %0, %1, off"
               :: "v"(lds_off), "v"(g) : "memory");
#endif
}

__device__ inline void async_wait0() {
#if __has_builtin(__builtin_amdgcn_s_wait_asynccnt)
  __builtin_amdgcn_s_wait_asynccnt(0);
#else
  asm volatile("s_wait_asynccnt 0" ::: "memory");
#endif
}

// ----------------------------- weight prep ---------------------------------
// in: f32, batch slabs of Kd x Nd (row major).  out: bf16, Nd x Kd per slab.
__global__ __launch_bounds__(256) void k_tr(const float* __restrict__ in,
                                            bf16* __restrict__ out,
                                            int Kd, int Nd, long total) {
  long slab = (long)Kd * Nd;
  for (long idx = (long)blockIdx.x * 256 + threadIdx.x; idx < total;
       idx += (long)gridDim.x * 256) {
    long bi  = idx / slab;
    long rem = idx - bi * slab;
    long k   = rem / Nd;
    long n   = rem - k * Nd;
    out[bi * slab + n * (long)Kd + k] = (bf16)in[idx];
  }
}

// ------------------------- embedding + posenc -------------------------------
__global__ __launch_bounds__(256) void k_embed(const int* __restrict__ x,
                                               const float* __restrict__ emb,
                                               float* __restrict__ h,
                                               bf16* __restrict__ hb) {
  long row = blockIdx.x;            // over B*S
  int  s   = (int)(row % Sq);
  long tok = x[row];
  const float ln1e4 = 9.210340371976184f;  // ln(10000)
#pragma unroll
  for (int i = 0; i < 4; ++i) {
    int   j   = threadIdx.x + i * 256;       // 0..D-1
    int   ii  = j >> 1;
    float ang = (float)s * __expf(-(2.0f * ii / (float)Dq) * ln1e4);
    float pe  = (j & 1) ? __cosf(ang) : __sinf(ang);
    float v   = emb[tok * (long)Dq + j] + pe;
    h[row * (long)Dq + j]  = v;
    hb[row * (long)Dq + j] = (bf16)v;
  }
}

// ----------------------------- WMMA GEMM ------------------------------------
// C[m,n] = scale * sum_k A[m,k]*Bt[n,k] + bias[n]; batched over blockIdx.z
// decomposed as (hh, bb).  256 threads = 8 waves; wave w -> rows
// blockIdx.y*128 + w*16, cols blockIdx.x*64 (shared across the block).
// The 64x32 bf16 B tile is async-DMA'd into LDS once per block per K-step
// (double buffered); all 8 waves read fragments from LDS.
__global__ __launch_bounds__(256) void k_gemm(
    const bf16* __restrict__ A,  long lda, long sAh, long sAb,
    const bf16* __restrict__ Bt, long ldb, long sBh, long sBb,
    const float* __restrict__ bias, long sBiasH,
    void* __restrict__ C, long ldc, long sCh, long sCb,
    int M, int N, int K, int Bdim, float scale, int flags) {
  (void)M; (void)N;                // launch grids are exact; no tails
  __shared__ bf16 smB[2][64][32];  // [buf][n_local][k_local] : 2 x 4KB

  const int lane  = threadIdx.x & 31;
  const int wave  = threadIdx.x >> 5;
  const int half  = lane >> 4;    // lane group 0/1
  const int lr    = lane & 15;
  const int tileM = blockIdx.y * 128 + wave * 16;
  const int tileN = blockIdx.x * 64;

  const int hh = blockIdx.z / Bdim;
  const int bb = blockIdx.z % Bdim;
  A  += (size_t)hh * sAh + (size_t)bb * sAb;
  Bt += (size_t)hh * sBh + (size_t)bb * sBb;
  const size_t coff = (size_t)hh * sCh + (size_t)bb * sCb;

  // A fragment rows: lane lr holds row M=tileM+lr.
  const bf16* arow = A + (size_t)(tileM + lr) * lda;

  // Cooperative B staging: thread t copies 16B: row tn, chunk tc.
  const int tn = threadIdx.x >> 2;   // 0..63 (local column of C)
  const int tc = threadIdx.x & 3;    // 0..3  (16B chunk within 32 k-values)
  const bf16* bsrc = Bt + (size_t)(tileN + tn) * ldb + tc * 8;
  bf16* bdst[2] = { &smB[0][tn][tc * 8], &smB[1][tn][tc * 8] };

  v8f acc[4];
#pragma unroll
  for (int j = 0; j < 4; ++j)
#pragma unroll
    for (int r = 0; r < 8; ++r) acc[j][r] = 0.0f;

  async_copy_b128(bsrc, bdst[0]);    // prologue: buffer 0 <- k0 = 0
  int cur = 0;
  for (int k0 = 0; k0 < K; k0 += 32) {
    async_wait0();                   // my 16B of smB[cur] has landed
    __syncthreads();                 // everyone's has landed
    if (k0 + 32 < K)                 // overlap next tile's DMA with compute
      async_copy_b128(bsrc + k0 + 32, bdst[cur ^ 1]);
    if (k0 + 64 <= K)                // keep the A stream warm
      __builtin_prefetch(arow + k0 + 64, 0, 0);

    // 16-bit A 16x32 layout: lane group 0 -> K {0..7,16..23}, group 1 -> +8.
    v8bf alo = *(const v8bf*)(arow + k0 + half * 8);
    v8bf ahi = *(const v8bf*)(arow + k0 + 16 + half * 8);
    v16bf af;
#pragma unroll
    for (int i = 0; i < 8; ++i) { af[i] = alo[i]; af[i + 8] = ahi[i]; }

#pragma unroll
    for (int j = 0; j < 4; ++j) {
      // 16-bit B 32x16 layout: lane group 0 -> K {0..7,16..23}, group 1 -> +8.
      const bf16* bp = &smB[cur][j * 16 + lr][half * 8];
      v8bf b0 = *(const v8bf*)bp;
      v8bf b1 = *(const v8bf*)(bp + 16);
      v16bf bfg;
#pragma unroll
      for (int i = 0; i < 8; ++i) { bfg[i] = b0[i]; bfg[i + 8] = b1[i]; }
      acc[j] = __builtin_amdgcn_wmma_f32_16x16x32_bf16(
          false, af, false, bfg, (short)0, acc[j], false, false);
    }
    cur ^= 1;
  }

#pragma unroll
  for (int j = 0; j < 4; ++j) {
    const int col = tileN + j * 16 + lr;
    float bv = 0.0f;
    if (bias) bv = bias[(size_t)hh * sBiasH + col];
#pragma unroll
    for (int r = 0; r < 8; ++r) {
      float v = acc[j][r] * scale + bv;
      if (flags & FLAG_RELU) v = fmaxf(v, 0.0f);
      const int row = tileM + half * 8 + r;   // C layout: VGPR r -> M=r(+8)
      size_t off = (flags & FLAG_TRANS)
                       ? coff + (size_t)col * ldc + row
                       : coff + (size_t)row * ldc + col;
      if (flags & FLAG_BF16) ((bf16*)C)[off]  = (bf16)v;
      else                   ((float*)C)[off] = v;
    }
  }
}

// ------------------------- causal softmax -----------------------------------
// scores f32 [H,B,S,S] -> attn bf16, one block per row.
__global__ __launch_bounds__(256) void k_softmax(const float* __restrict__ sc,
                                                 bf16* __restrict__ attn) {
  __shared__ float red[256];
  const long row = blockIdx.x;      // over H*B*S
  const int  s   = (int)(row % Sq); // query position: keep cols <= s
  const float* p = sc + row * (long)Sq;
  bf16*        q = attn + row * (long)Sq;
  const int t = threadIdx.x;
  float vals[4];
  float mx = -3.0e38f;
#pragma unroll
  for (int i = 0; i < 4; ++i) {
    int j   = t + i * 256;
    float v = (j <= s) ? p[j] : -3.0e38f;
    vals[i] = v;
    mx      = fmaxf(mx, v);
  }
  red[t] = mx; __syncthreads();
  for (int o = 128; o > 0; o >>= 1) {
    if (t < o) red[t] = fmaxf(red[t], red[t + o]);
    __syncthreads();
  }
  mx = red[0]; __syncthreads();
  float sum = 0.0f;
#pragma unroll
  for (int i = 0; i < 4; ++i) {
    int j   = t + i * 256;
    vals[i] = (j <= s) ? __expf(vals[i] - mx) : 0.0f;
    sum += vals[i];
  }
  red[t] = sum; __syncthreads();
  for (int o = 128; o > 0; o >>= 1) {
    if (t < o) red[t] += red[t + o];
    __syncthreads();
  }
  const float inv = 1.0f / red[0];
#pragma unroll
  for (int i = 0; i < 4; ++i) q[t + i * 256] = (bf16)(vals[i] * inv);
}

// ------------------- residual add + LayerNorm -------------------------------
__global__ __launch_bounds__(256) void k_ln_add(const float* __restrict__ a,
                                                const float* __restrict__ res,
                                                const float* __restrict__ g,
                                                const float* __restrict__ be,
                                                float* __restrict__ outf,
                                                bf16* __restrict__ outb) {
  __shared__ float red[256];
  const long base = blockIdx.x * (long)Dq;
  const int  t    = threadIdx.x;
  float v[4], s = 0.0f;
#pragma unroll
  for (int i = 0; i < 4; ++i) {
    int j = t + i * 256;
    v[i]  = a[base + j] + res[base + j];
    s += v[i];
  }
  red[t] = s; __syncthreads();
  for (int o = 128; o > 0; o >>= 1) { if (t < o) red[t] += red[t + o]; __syncthreads(); }
  const float mu = red[0] * (1.0f / Dq); __syncthreads();
  float s2 = 0.0f;
#pragma unroll
  for (int i = 0; i < 4; ++i) { float d = v[i] - mu; s2 += d * d; }
  red[t] = s2; __syncthreads();
  for (int o = 128; o > 0; o >>= 1) { if (t < o) red[t] += red[t + o]; __syncthreads(); }
  const float inv = rsqrtf(red[0] * (1.0f / Dq) + 1e-5f);
#pragma unroll
  for (int i = 0; i < 4; ++i) {
    int   j = t + i * 256;
    float y = (v[i] - mu) * inv * g[j] + be[j];
    outf[base + j] = y;
    outb[base + j] = (bf16)y;
  }
}

// ----------------------------- host side ------------------------------------
extern "C" void kernel_launch(void* const* d_in, const int* in_sizes, int n_in,
                              void* d_out, int out_size, void* d_ws, size_t ws_size,
                              hipStream_t stream) {
  (void)in_sizes; (void)n_in; (void)out_size; (void)ws_size;
  const int*   x     = (const int*)d_in[0];
  const float* emb   = (const float*)d_in[1];
  const float* wq    = (const float*)d_in[2];
  const float* bq    = (const float*)d_in[3];
  const float* wk    = (const float*)d_in[4];
  const float* bk    = (const float*)d_in[5];
  const float* wv    = (const float*)d_in[6];
  const float* bv    = (const float*)d_in[7];
  const float* wr    = (const float*)d_in[8];
  const float* br    = (const float*)d_in[9];
  const float* ln1g  = (const float*)d_in[10];
  const float* ln1b  = (const float*)d_in[11];
  const float* w1    = (const float*)d_in[12];
  const float* b1    = (const float*)d_in[13];
  const float* w2    = (const float*)d_in[14];
  const float* b2    = (const float*)d_in[15];
  const float* ln2g  = (const float*)d_in[16];
  const float* ln2b  = (const float*)d_in[17];
  const float* wf    = (const float*)d_in[18];
  const float* bff   = (const float*)d_in[19];
  float* out = (float*)d_out;

  // ---- workspace carving (256B aligned) ----
  size_t cur = 0;
  auto carve = [&](size_t bytes) {
    size_t o = (cur + 255) & ~(size_t)255;
    cur = o + bytes;
    return (char*)d_ws + o;
  };
  const long HE = (long)Hq * Eq;                      // 8192
  bf16*  wq_t = (bf16*)carve((size_t)Lq * Hq * Eq * Dq * 2);
  bf16*  wk_t = (bf16*)carve((size_t)Lq * Hq * Eq * Dq * 2);
  bf16*  wv_t = (bf16*)carve((size_t)Lq * Hq * Eq * Dq * 2);
  bf16*  wr_t = (bf16*)carve((size_t)Lq * Dq * HE * 2);
  bf16*  w1_t = (bf16*)carve((size_t)Lq * Fq * Dq * 2);
  bf16*  w2_t = (bf16*)carve((size_t)Lq * Dq * Fq * 2);
  bf16*  wf_t = (bf16*)carve((size_t)Vq * Dq * 2);
  float* h    = (float*)carve((size_t)Bq * Sq * Dq * 4);
  bf16*  hb   = (bf16*)carve((size_t)Bq * Sq * Dq * 2);
  bf16*  Qb   = (bf16*)carve((size_t)Hq * Bq * Sq * Eq * 2);  // also: attn
  bf16*  Kb   = (bf16*)carve((size_t)Hq * Bq * Sq * Eq * 2);  // also: FFN t
  bf16*  Vt   = (bf16*)carve((size_t)Hq * Bq * Eq * Sq * 2);
  float* scr  = (float*)carve((size_t)Hq * Bq * Sq * Sq * 4); // also: o_bf
  float* af   = (float*)carve((size_t)Bq * Sq * Dq * 4);      // also: ff
  float* x1   = (float*)carve((size_t)Bq * Sq * Dq * 4);
  bf16*  x1b  = (bf16*)carve((size_t)Bq * Sq * Dq * 2);
  bf16*  attn = Qb;          // Q dead after scores GEMM
  bf16*  tb   = Kb;          // K dead after scores GEMM (33.5MB fits in 67MB)
  bf16*  ob   = (bf16*)scr;  // scores dead after softmax
  float* ff   = af;          // a dead after first LayerNorm

  auto gemm = [&](const bf16* A, long lda, long sAh, long sAb,
                  const bf16* Bt, long ldb, long sBh, long sBb,
                  const float* bias, long sBiasH,
                  void* C, long ldc, long sCh, long sCb,
                  int M, int N, int K, int Hdim, int Bdim, float scale, int flags) {
    dim3 grid(N / 64, (M + 127) / 128, Hdim * Bdim);
    k_gemm<<<grid, 256, 0, stream>>>(A, lda, sAh, sAb, Bt, ldb, sBh, sBb,
                                     bias, sBiasH, C, ldc, sCh, sCb,
                                     M, N, K, Bdim, scale, flags);
  };

  // ---- weight conversion: f32 [.,K,N] -> bf16 [.,N,K] ----
  const int TB = 8192;
  k_tr<<<TB, 256, 0, stream>>>(wq, wq_t, Dq, Eq, (long)Lq * Hq * Dq * Eq);
  k_tr<<<TB, 256, 0, stream>>>(wk, wk_t, Dq, Eq, (long)Lq * Hq * Dq * Eq);
  k_tr<<<TB, 256, 0, stream>>>(wv, wv_t, Dq, Eq, (long)Lq * Hq * Dq * Eq);
  k_tr<<<TB, 256, 0, stream>>>(wr, wr_t, (int)HE, Dq, (long)Lq * HE * Dq);
  k_tr<<<TB, 256, 0, stream>>>(w1, w1_t, Dq, Fq, (long)Lq * Dq * Fq);
  k_tr<<<TB, 256, 0, stream>>>(w2, w2_t, Fq, Dq, (long)Lq * Fq * Dq);
  k_tr<<<TB, 256, 0, stream>>>(wf, wf_t, Dq, Vq, (long)Dq * Vq);

  // ---- embedding + positional encoding ----
  k_embed<<<Bq * Sq, 256, 0, stream>>>(x, emb, h, hb);

  for (int l = 0; l < Lq; ++l) {
    const bf16* wq_l = wq_t + (size_t)l * Hq * Eq * Dq;
    const bf16* wk_l = wk_t + (size_t)l * Hq * Eq * Dq;
    const bf16* wv_l = wv_t + (size_t)l * Hq * Eq * Dq;
    // Q,K: bf16 [H,B,S,E]; V: bf16 transposed [H,B,E,S]
    gemm(hb, Dq, 0, (long)Sq * Dq, wq_l, Dq, (long)Eq * Dq, 0,
         bq + (size_t)l * Hq * Eq, Eq,
         Qb, Eq, (long)Bq * Sq * Eq, (long)Sq * Eq,
         Sq, Eq, Dq, Hq, Bq, 1.0f, FLAG_BF16);
    gemm(hb, Dq, 0, (long)Sq * Dq, wk_l, Dq, (long)Eq * Dq, 0,
         bk + (size_t)l * Hq * Eq, Eq,
         Kb, Eq, (long)Bq * Sq * Eq, (long)Sq * Eq,
         Sq, Eq, Dq, Hq, Bq, 1.0f, FLAG_BF16);
    gemm(hb, Dq, 0, (long)Sq * Dq, wv_l, Dq, (long)Eq * Dq, 0,
         bv + (size_t)l * Hq * Eq, Eq,
         Vt, Sq, (long)Bq * Eq * Sq, (long)Eq * Sq,
         Sq, Eq, Dq, Hq, Bq, 1.0f, FLAG_BF16 | FLAG_TRANS);
    // scores = Q K^T / sqrt(D): f32 [H,B,S,S]
    gemm(Qb, Eq, (long)Bq * Sq * Eq, (long)Sq * Eq,
         Kb, Eq, (long)Bq * Sq * Eq, (long)Sq * Eq, nullptr, 0,
         scr, Sq, (long)Bq * Sq * Sq, (long)Sq * Sq,
         Sq, Sq, Eq, Hq, Bq, 0.03125f, 0);
    // causal softmax -> attn bf16 (aliases Qb)
    k_softmax<<<Hq * Bq * Sq, 256, 0, stream>>>(scr, attn);
    // o = attn @ V, stored concat [B,S,H*E] bf16 (aliases scores)
    gemm(attn, Sq, (long)Bq * Sq * Sq, (long)Sq * Sq,
         Vt, Sq, (long)Bq * Eq * Sq, (long)Eq * Sq, nullptr, 0,
         ob, HE, (long)Eq, (long)Sq * HE,
         Sq, Eq, Sq, Hq, Bq, 1.0f, FLAG_BF16);
    // a = o @ wr + br : f32 [B*S, D]
    gemm(ob, HE, 0, 0, wr_t + (size_t)l * Dq * HE, HE, 0, 0,
         br + (size_t)l * Dq, 0, af, Dq, 0, 0,
         Bq * Sq, Dq, (int)HE, 1, 1, 1.0f, 0);
    // ff_in = LN(a + h)
    k_ln_add<<<Bq * Sq, 256, 0, stream>>>(af, h, ln1g + (size_t)l * Dq,
                                          ln1b + (size_t)l * Dq, x1, x1b);
    // t = relu(ff_in @ w1 + b1) bf16 (aliases Kb)
    gemm(x1b, Dq, 0, 0, w1_t + (size_t)l * Fq * Dq, Dq, 0, 0,
         b1 + (size_t)l * Fq, 0, tb, Fq, 0, 0,
         Bq * Sq, Fq, Dq, 1, 1, 1.0f, FLAG_BF16 | FLAG_RELU);
    // ff = t @ w2 + b2 : f32 (aliases af)
    gemm(tb, Fq, 0, 0, w2_t + (size_t)l * Dq * Fq, Fq, 0, 0,
         b2 + (size_t)l * Dq, 0, ff, Dq, 0, 0,
         Bq * Sq, Dq, Fq, 1, 1, 1.0f, 0);
    // h = LN(ff + ff_in)
    k_ln_add<<<Bq * Sq, 256, 0, stream>>>(ff, x1, ln2g + (size_t)l * Dq,
                                          ln2b + (size_t)l * Dq, h, hb);
  }

  // logits = h @ wf + bf : f32 [B*S, V] straight to d_out
  gemm(hb, Dq, 0, 0, wf_t, Dq, 0, 0, bff, 0,
       out, Vq, 0, 0, Bq * Sq, Vq, Dq, 1, 1, 1.0f, 0);
}
